// PosConcatMesPassing_64046552318005
// MI455X (gfx1250) — matrix operand order repaired and checked by reading, not compiled, not run on hardware
//
#include <hip/hip_runtime.h>
#include <hip/hip_bf16.h>
#include <math.h>

typedef __attribute__((ext_vector_type(16))) _Float16 v16h;
typedef __attribute__((ext_vector_type(8)))  float    v8f;
typedef __attribute__((ext_vector_type(4)))  float    f4v;

#define NDIM    128
#define EDIM    64
#define PDIM    16
#define CDIM    128
#define INDIM   336
#define KPAD    352          // 336 padded to multiple of 32
#define KSTEPS  11           // KPAD / 32
#define NNODES  2048
#define NEDGES  32768
#define MAXE    2048         // per-node incident-edge list cap (avg ~16)

// ---- WMMA A-fragment loader (CDNA5 ISA 7.12.2 layout, wave32) ----
// A: 16x32 f16, lane l holds row M=l%16; lanes 0-15: K in {0..7,16..23},
// lanes 16-31: K in {8..15,24..31} -> two contiguous 16B chunks 32B apart,
// which the compiler lowers to two ds_load_b128.
__device__ __forceinline__ v16h load_a_frag(const _Float16* xs, int lane, int k0) {
  const int m  = lane & 15;
  const int kb = (lane < 16) ? 0 : 8;
  const _Float16* rowp = xs + m * KPAD;
  v16h a;
#pragma unroll
  for (int vv = 0; vv < 8; ++vv) {
    int k = k0 + kb + ((vv < 4) ? (2 * vv) : (16 + 2 * (vv - 4)));
    a[2 * vv]     = rowp[k];
    a[2 * vv + 1] = rowp[k + 1];
  }
  return a;
}

__device__ __forceinline__ float leaky(float x) { return (x >= 0.0f) ? x : 0.01f * x; }

// -------- Kernel 0: pack fp32 weights into f16 WMMA B-fragment order --------
// B: 32x16 (K x N), B[k][n] = W[n][k]. Lane l holds column n=l%16;
// lanes 0-15: K in {0..7,16..23}, lanes 16-31: K in {8..15,24..31}, as 8 K-pairs.
// Packed record: dst[((ntile*KSTEPS + kstep)*32 + lane)] = v16h (32 bytes).
__global__ __launch_bounds__(256)
void pack_weights_kernel(const float* __restrict__ We, const float* __restrict__ W,
                         v16h* __restrict__ WeP, v16h* __restrict__ WP) {
  const int nWe = 4 * KSTEPS * 32;   // 64 output cols  -> 4 N-tiles
  const int nW  = 8 * KSTEPS * 32;   // 128 output cols -> 8 N-tiles
  int idx = blockIdx.x * 256 + threadIdx.x;
  if (idx >= nWe + nW) return;
  const float* src; v16h* dst; int lidx;
  if (idx < nWe) { src = We; dst = WeP; lidx = idx; }
  else           { src = W;  dst = WP;  lidx = idx - nWe; }
  const int lane  = lidx & 31;
  const int kstep = (lidx >> 5) % KSTEPS;
  const int ntile = (lidx >> 5) / KSTEPS;
  const int n  = lane & 15;
  const int kb = (lane < 16) ? 0 : 8;
  const int k0 = kstep * 32;
  const float* wr = src + (size_t)(ntile * 16 + n) * INDIM;
  v16h b;
#pragma unroll
  for (int vv = 0; vv < 8; ++vv) {
    int k = k0 + kb + ((vv < 4) ? (2 * vv) : (16 + 2 * (vv - 4)));
    b[2 * vv]     = (_Float16)((k     < INDIM) ? wr[k]     : 0.0f);
    b[2 * vv + 1] = (_Float16)((k + 1 < INDIM) ? wr[k + 1] : 0.0f);
  }
  dst[lidx] = b;
}

// ---------------- Kernel 1: fused edge MLP (WMMA) ----------------
// Per block: 16 edges. Computes new_edge [E,64] (global out + LDS x2),
// neighbor [E,128] (workspace), attention logits a [E] (workspace).
__global__ __launch_bounds__(256)
void edge_mlp_kernel(const float* __restrict__ u,  const float* __restrict__ v,
                     const float* __restrict__ ef, const float* __restrict__ pf,
                     const v16h* __restrict__ WeP, const float* __restrict__ be,
                     const v16h* __restrict__ WP,  const float* __restrict__ b,
                     const float* __restrict__ Wa, const float* __restrict__ ba,
                     float* __restrict__ newedge_out,
                     float* __restrict__ neighbor_out,
                     float* __restrict__ a_out) {
  __shared__ _Float16 xs[16 * KPAD];   // 11 KB: x1 tile, then becomes x2 tile
  __shared__ float red[256];

  const int tid  = threadIdx.x;
  const int lane = tid & 31;
  const int wid  = tid >> 5;
  const int e0   = blockIdx.x * 16;

  // ---- stage x1 = [u | edge | pos | v | 0-pad] into LDS as f16 ----
  for (int idx = tid; idx < 16 * KPAD; idx += 256) {
    int r = idx / KPAD, c = idx - r * KPAD;
    int e = e0 + r;
    float val;
    if (c < 128)       val = u[(size_t)e * 128 + c];
    else if (c < 192)  val = ef[(size_t)e * 64 + (c - 128)];
    else if (c < 208)  val = pf[(size_t)e * 16 + (c - 192)];
    else if (c < 336)  val = v[(size_t)e * 128 + (c - 208)];
    else               val = 0.0f;
    xs[idx] = (_Float16)val;
  }
  __syncthreads();

  // ---- phase B: new_edge = leaky(x1 @ We^T + be), waves 0..3 ----
  v8f accE = {};
  if (wid < 4) {
    const v16h* bp = WeP + (size_t)wid * KSTEPS * 32 + lane;
    for (int ks = 0; ks < KSTEPS; ++ks) {
      v16h A = load_a_frag(xs, lane, ks * 32);
      v16h B = bp[ks * 32];
      accE = __builtin_amdgcn_wmma_f32_16x16x32_f16(false, A, false, B,
                                                    (short)0, accE, false, false);
    }
  }
  __syncthreads();   // all reads of x1 done before overwriting edge cols
  if (wid < 4) {
    const int n0 = wid * 16;
    const int col = lane & 15;
    const int rbase = (lane < 16) ? 0 : 8;
#pragma unroll
    for (int j = 0; j < 8; ++j) {
      int r = rbase + j;
      float val = leaky(accE[j] + be[n0 + col]);
      newedge_out[(size_t)(e0 + r) * 64 + n0 + col] = val;
      xs[r * KPAD + 128 + n0 + col] = (_Float16)val;   // form x2 in place
    }
  }
  __syncthreads();

  // ---- phase C: neighbor = leaky(x2 @ W^T + b), all 8 waves ----
  {
    v8f accN = {};
    const v16h* bp = WP + (size_t)wid * KSTEPS * 32 + lane;
    for (int ks = 0; ks < KSTEPS; ++ks) {
      v16h A = load_a_frag(xs, lane, ks * 32);
      v16h B = bp[ks * 32];
      accN = __builtin_amdgcn_wmma_f32_16x16x32_f16(false, A, false, B,
                                                    (short)0, accN, false, false);
    }
    const int n0 = wid * 16;
    const int col = lane & 15;
    const int rbase = (lane < 16) ? 0 : 8;
#pragma unroll
    for (int j = 0; j < 8; ++j) {
      int r = rbase + j;
      float val = leaky(accN[j] + b[n0 + col]);
      neighbor_out[(size_t)(e0 + r) * 128 + n0 + col] = val;
    }
  }

  // ---- phase D: attention logits a = leaky(x2 @ Wa^T + ba) ----
  {
    const int r   = tid >> 4;   // edge row 0..15
    const int seg = tid & 15;
    float s = 0.0f;
    for (int k = seg; k < INDIM; k += 16)
      s += (float)xs[r * KPAD + k] * Wa[k];
    red[tid] = s;
    __syncthreads();
#pragma unroll
    for (int off = 8; off > 0; off >>= 1) {
      if (seg < off) red[tid] += red[tid + off];
      __syncthreads();
    }
    if (seg == 0)
      a_out[e0 + r] = leaky(red[tid] + ba[0]);
  }
}

// ---------------- Kernel 2: sparse masked softmax + aggregation ----------------
// One block per node. Streams the node's incidence row (non-temporal),
// collects incident edges, exact softmax (non-incident terms underflow to 0
// exactly as in the dense reference), gathers neighbor rows, applies ELU.
__global__ __launch_bounds__(256)
void aggregate_kernel(const float* __restrict__ M, const float* __restrict__ a,
                      const float* __restrict__ neighbor, float* __restrict__ ctx) {
  __shared__ int   cnt;
  __shared__ int   idxs[MAXE];
  __shared__ float wsh[MAXE];
  __shared__ float invsum_sh;

  const int tid = threadIdx.x;
  const int n   = blockIdx.x;
  if (tid == 0) cnt = 0;
  __syncthreads();

  const f4v* row4 = (const f4v*)(M + (size_t)n * NEDGES);
  for (int q = tid; q < NEDGES / 4; q += 256) {
    f4v m4 = __builtin_nontemporal_load(row4 + q);
#pragma unroll
    for (int j = 0; j < 4; ++j) {
      if (m4[j] != 0.0f) {
        int p = atomicAdd(&cnt, 1);
        if (p < MAXE) idxs[p] = q * 4 + j;
      }
    }
  }
  __syncthreads();

  int c = cnt; if (c > MAXE) c = MAXE;

  if (c == 0) {
    // row fully masked: softmax is uniform over all 32768 edges
    if (tid < 128) {
      float s = 0.0f;
      for (int e = 0; e < NEDGES; ++e) s += neighbor[(size_t)e * 128 + tid];
      s *= (1.0f / (float)NEDGES);
      ctx[(size_t)n * 128 + tid] = (s > 0.0f) ? s : expm1f(s);
    }
    return;
  }

  for (int i = tid; i < c; i += 256) wsh[i] = a[idxs[i]];
  __syncthreads();

  if (tid == 0) {           // c is ~16 on average; serial is fine
    float mx = -3.0e38f;
    for (int i = 0; i < c; ++i) mx = fmaxf(mx, wsh[i]);
    float sum = 0.0f;
    for (int i = 0; i < c; ++i) { float ev = expf(wsh[i] - mx); wsh[i] = ev; sum += ev; }
    invsum_sh = 1.0f / sum;
  }
  __syncthreads();

  if (tid < 128) {
    const float inv = invsum_sh;
    float s = 0.0f;
    for (int i = 0; i < c; ++i)
      s += wsh[i] * neighbor[(size_t)idxs[i] * 128 + tid];
    s *= inv;
    ctx[(size_t)n * 128 + tid] = (s > 0.0f) ? s : expm1f(s);   // ELU
  }
}

extern "C" void kernel_launch(void* const* d_in, const int* in_sizes, int n_in,
                              void* d_out, int out_size, void* d_ws, size_t ws_size,
                              hipStream_t stream) {
  const float* u  = (const float*)d_in[0];
  const float* v  = (const float*)d_in[1];
  const float* ef = (const float*)d_in[2];
  const float* pf = (const float*)d_in[3];
  const float* M  = (const float*)d_in[4];
  // d_in[5] (node_edge_mask) is mathematically (1-M)*(-1e4): never read.
  const float* We = (const float*)d_in[6];
  const float* be = (const float*)d_in[7];
  const float* W  = (const float*)d_in[8];
  const float* b  = (const float*)d_in[9];
  const float* Wa = (const float*)d_in[10];
  const float* ba = (const float*)d_in[11];

  float* ctx      = (float*)d_out;                          // [2048,128]
  float* newedge  = (float*)d_out + (size_t)NNODES * CDIM;  // [32768,64]

  // workspace layout (all 32B-aligned offsets)
  char* ws = (char*)d_ws;
  float* neighbor = (float*)ws;                                   // 32768*128 f32 = 16 MB
  float* a_ws     = (float*)(ws + (size_t)NEDGES * CDIM * 4);     // 32768 f32
  v16h*  WeP      = (v16h*)(ws + (size_t)NEDGES * CDIM * 4 + (size_t)NEDGES * 4);
  v16h*  WP       = WeP + (size_t)4 * KSTEPS * 32;                // after 4 N-tiles of WeP

  const int packThreads = (4 + 8) * KSTEPS * 32;                  // 4224
  pack_weights_kernel<<<(packThreads + 255) / 256, 256, 0, stream>>>(We, W, WeP, WP);
  edge_mlp_kernel<<<NEDGES / 16, 256, 0, stream>>>(u, v, ef, pf, WeP, be, WP, b,
                                                   Wa, ba, newedge, neighbor, a_ws);
  aggregate_kernel<<<NNODES, 256, 0, stream>>>(M, a_ws, neighbor, ctx);
}
